// FrameSimilarity_49976239456335
// MI455X (gfx1250) — matrix operand order repaired and checked by reading, not compiled, not run on hardware
//
#include <hip/hip_runtime.h>
#include <hip/hip_bf16.h>

// Problem constants
#define TB    2
#define TC    512
#define TT    1024
#define THW   49
#define TIN   1024   // C0 + C1
#define TSIM  128
#define TLOOK 101
#define TOUT  128

#define POOL_TCHUNK 256                       // t's per pooling block
#define POOL_ELEMS  (POOL_TCHUNK * THW)       // 12544 floats = 49 KiB

typedef __attribute__((ext_vector_type(4)))  float          floatx4;
typedef __attribute__((ext_vector_type(8)))  float          floatx8;
typedef __attribute__((ext_vector_type(8)))  unsigned short u16x8;
typedef __attribute__((ext_vector_type(16))) unsigned short u16x16;
typedef __attribute__((ext_vector_type(16))) __bf16         bf16x16;
typedef __attribute__((ext_vector_type(4)))  unsigned int   uint32x4;
typedef __attribute__((ext_vector_type(8)))  int            int32x8;
typedef __attribute__((ext_vector_type(4)))  int            int32x4;

#if defined(__gfx1250__) && __has_builtin(__builtin_amdgcn_tensor_load_to_lds) && \
    __has_builtin(__builtin_amdgcn_s_wait_tensorcnt)
#define USE_TDM 1
#else
#define USE_TDM 0
#endif

__device__ __forceinline__ unsigned short f2bf(float f) {
    unsigned int b = __float_as_uint(f);
    b += 0x7FFFu + ((b >> 16) & 1u);           // round-to-nearest-even
    return (unsigned short)(b >> 16);
}

union BFrag { u16x16 u; bf16x16 b; };

// A-matrix fragment (16x32 bf16): lane m = lane%16, half = lane/16.
// elements e = K(kk + 16*half + e), e = 0..15
__device__ __forceinline__ bf16x16 load_a_frag(const unsigned short* row, int kk, int hh) {
    int k0 = kk + 16 * hh;
    u16x8 lo = *(const u16x8*)(row + k0);
    u16x8 hi = *(const u16x8*)(row + k0 + 8);
    BFrag f;
    f.u = __builtin_shufflevector(lo, hi, 0,1,2,3,4,5,6,7,8,9,10,11,12,13,14,15);
    return f.b;
}

// B-matrix fragment (32x16 bf16), from a row-major BT[n][k] buffer:
// lane n = lane%16, elements e = K(kk + 16*half + e), contiguous 32B
__device__ __forceinline__ bf16x16 load_b_frag(const unsigned short* rowT, int kk, int hh) {
    BFrag f;
    f.u = *(const u16x16*)(rowT + kk + 16 * hh);
    return f.b;
}

// ---------------------------------------------------------------------------
// K0: convert weights to bf16, transposed for B-fragment loads.
//   projw_t[n][k] = bf16(proj_w[k][n])            (128 x 1024)
//   fcw_t  [n][k] = bf16(fc_w[k][n]) k<101 else 0 (128 x 128, K padded)
// ---------------------------------------------------------------------------
__global__ void k0_convert(const float* __restrict__ projw, const float* __restrict__ fcw,
                           unsigned short* __restrict__ projw_t, unsigned short* __restrict__ fcw_t) {
    int idx = blockIdx.x * blockDim.x + threadIdx.x;
    if (idx < TIN * TSIM) {
        int k = idx / TSIM, n = idx % TSIM;
        projw_t[n * TIN + k] = f2bf(projw[idx]);
    } else {
        int i2 = idx - TIN * TSIM;
        if (i2 < TOUT * 128) {
            int n = i2 / 128, k = i2 % 128;
            float v = (k < TLOOK) ? fcw[k * TOUT + n] : 0.0f;
            fcw_t[n * 128 + k] = f2bf(v);
        }
    }
}

// ---------------------------------------------------------------------------
// K1: HBM-bound pooling.  One block per (b, channel) plane x 256-t chunk.
// The 256*49 floats per chunk are contiguous: move them into LDS either with
// one Tensor-Data-Mover DMA (TDM, TENSORcnt-tracked, zero VGPR staging) or a
// coalesced float4 copy fallback.  Then 256 lanes each sum their 49 values
// (stride 49 vs 64 banks -> conflict-free) and write bf16 feats[b][t][c].
// ---------------------------------------------------------------------------
__global__ void k1_pool(const float* __restrict__ x0, const float* __restrict__ x1,
                        unsigned short* __restrict__ feats) {
    __shared__ __align__(16) float lds[POOL_ELEMS];
    int plane = blockIdx.x;                 // b*1024 + c
    int b = plane >> 10;
    int c = plane & 1023;
    int t0 = blockIdx.y * POOL_TCHUNK;
    const float* src = (c < TC)
        ? (x0 + ((size_t)(b * TC + c)        * TT + t0) * THW)
        : (x1 + ((size_t)(b * TC + (c - TC)) * TT + t0) * THW);

#if USE_TDM
    if (threadIdx.x < 32) {                 // wave 0 only: TDM ignores EXEC,
                                            // duplicate issue would re-DMA
        unsigned long long ga = (unsigned long long)(uintptr_t)src;
        unsigned lds_off = (unsigned)(uintptr_t)&lds[0];   // low 32b of generic
                                                           // LDS ptr = LDS offset
        // D# group0: count=1 | lds_addr | global_addr[56:0] | type=2
        uint32x4 g0 = { 1u,
                        lds_off,
                        (unsigned)(ga & 0xFFFFFFFFu),
                        (unsigned)((ga >> 32) & 0x01FFFFFFu) | (2u << 30) };
        // D# group1: data_size=2 (4B); tensor_dim0 = tile_dim0 = 12544;
        //            tensor_dim1 = tile_dim1 = 1; dim0_stride = 12544
        int32x8 g1 = { (int)0x00020000u,                       // data_size=4B
                       (int)((POOL_ELEMS & 0xFFFFu) << 16),    // tensor_dim0 lo
                       (int)(((POOL_ELEMS >> 16) & 0xFFFFu) |  // tensor_dim0 hi
                             (1u << 16)),                      // tensor_dim1 = 1
                       (int)((POOL_ELEMS & 0xFFFFu) << 16),    // tile_dim0
                       1,                                      // tile_dim1 = 1
                       POOL_ELEMS,                             // dim0_stride lo
                       0, 0 };
        int32x4 g2 = { 0, 0, 0, 0 };
        int32x4 g3 = { 0, 0, 0, 0 };
        int32x8 g4 = { 0, 0, 0, 0, 0, 0, 0, 0 };
        __builtin_amdgcn_tensor_load_to_lds(g0, g1, g2, g3, g4, 0);
        __builtin_amdgcn_s_wait_tensorcnt(0);
    }
    __syncthreads();
#else
    const floatx4* src4 = (const floatx4*)src;
    for (int i = threadIdx.x; i < POOL_ELEMS / 4; i += blockDim.x)
        ((floatx4*)lds)[i] = src4[i];
    __syncthreads();
#endif

    int t = threadIdx.x;                    // 256 threads, 256 t's
    float s = 0.f;
    #pragma unroll
    for (int j = 0; j < THW; ++j) s += lds[t * THW + j];
    feats[(size_t)(b * TT + t0 + t) * TIN + c] = f2bf(s * (1.0f / 49.0f));
}

// ---------------------------------------------------------------------------
// K2: projection GEMM (16 rows x 128 cols per block; wave w owns N-tile w;
// K = 1024 in 32-steps of v_wmma_f32_16x16x32_bf16) + row L2 normalize.
// Output xn[b*T + t][d] as bf16.
// ---------------------------------------------------------------------------
__global__ void k2_proj_norm(const unsigned short* __restrict__ feats,
                             const unsigned short* __restrict__ projw_t,
                             unsigned short* __restrict__ xn) {
    __shared__ float p_lds[16][TSIM];
    __shared__ float s_scale[16];
    int r0   = blockIdx.x * 16;             // global row (b*T + t)
    int w    = threadIdx.x >> 5;
    int lane = threadIdx.x & 31;
    int hh   = lane >> 4;
    int ln   = lane & 15;

    const unsigned short* arow = feats   + (size_t)(r0 + ln) * TIN;
    const unsigned short* brow = projw_t + (size_t)(w * 16 + ln) * TIN;
    floatx8 acc = {};
    for (int kk = 0; kk < TIN; kk += 32) {
        bf16x16 a  = load_a_frag(arow, kk, hh);
        bf16x16 bm = load_b_frag(brow, kk, hh);
        acc = __builtin_amdgcn_wmma_f32_16x16x32_bf16(false, a, false, bm,
                                                      (short)0, acc, false, false);
    }
    #pragma unroll
    for (int r = 0; r < 8; ++r)
        p_lds[r + 8 * hh][w * 16 + ln] = acc[r];
    __syncthreads();
    if (threadIdx.x < 16) {
        float s = 0.f;
        for (int j = 0; j < TSIM; ++j) { float v = p_lds[threadIdx.x][j]; s += v * v; }
        s_scale[threadIdx.x] = 1.0f / fmaxf(sqrtf(s), 1e-12f);
    }
    __syncthreads();
    for (int i = threadIdx.x; i < 16 * TSIM; i += blockDim.x) {
        int r = i >> 7, cidx = i & 127;
        xn[(size_t)(r0 + r) * TSIM + cidx] = f2bf(p_lds[r][cidx] * s_scale[r]);
    }
}

// ---------------------------------------------------------------------------
// K3: fused banded similarity + fc + ReLU.  Per 16-t tile:
//  phase1: sim[16][128] = X_t (16x128) @ X_s^T, s in [t0-56, t0+72); OOB s -> 0
//  phase2: gather band: banded[t][j] = sim[t][(t-t0)+j+6], j<101 else 0 (bf16)
//  phase3: out = relu(banded @ fcw_t + bias)
// ---------------------------------------------------------------------------
__global__ void k3_sim_fc(const unsigned short* __restrict__ xn,
                          const unsigned short* __restrict__ fcw_t,
                          const float* __restrict__ fcb,
                          float* __restrict__ out) {
    __shared__ __align__(16) float          sim_lds[16][128];
    __shared__ __align__(16) unsigned short band_lds[16][128];
    int blk  = blockIdx.x;
    int b    = blk >> 6;                    // 64 tiles per batch
    int t0   = (blk & 63) * 16;
    int w    = threadIdx.x >> 5;
    int lane = threadIdx.x & 31;
    int hh   = lane >> 4;
    int ln   = lane & 15;

    // phase 1: similarity window GEMM
    {
        const unsigned short* arow = xn + (size_t)(b * TT + t0 + ln) * TSIM;
        int s = t0 - 56 + w * 16 + ln;
        bool valid = (s >= 0) && (s < TT);
        const unsigned short* brow = xn + (size_t)(b * TT + (valid ? s : 0)) * TSIM;
        floatx8 acc = {};
        for (int kk = 0; kk < TSIM; kk += 32) {
            bf16x16 a  = load_a_frag(arow, kk, hh);
            bf16x16 bm = {};
            if (valid) bm = load_b_frag(brow, kk, hh);
            acc = __builtin_amdgcn_wmma_f32_16x16x32_bf16(false, a, false, bm,
                                                          (short)0, acc, false, false);
        }
        #pragma unroll
        for (int r = 0; r < 8; ++r)
            sim_lds[r + 8 * hh][w * 16 + ln] = acc[r];
    }
    __syncthreads();

    // phase 2: band gather -> bf16 A-tile (K padded 101 -> 128)
    for (int i = threadIdx.x; i < 16 * 128; i += blockDim.x) {
        int r = i >> 7, j = i & 127;
        float v = (j < TLOOK) ? sim_lds[r][r + j + 6] : 0.0f;
        band_lds[r][j] = f2bf(v);
    }
    __syncthreads();

    // phase 3: banded @ fc_w + bias, relu
    {
        const unsigned short* arow = &band_lds[ln][0];
        const unsigned short* brow = fcw_t + (size_t)(w * 16 + ln) * 128;
        floatx8 acc = {};
        for (int kk = 0; kk < 128; kk += 32) {
            bf16x16 a  = load_a_frag(arow, kk, hh);
            bf16x16 bm = load_b_frag(brow, kk, hh);
            acc = __builtin_amdgcn_wmma_f32_16x16x32_bf16(false, a, false, bm,
                                                          (short)0, acc, false, false);
        }
        int n = w * 16 + ln;
        float bias = fcb[n];
        #pragma unroll
        for (int r = 0; r < 8; ++r) {
            int m = r + 8 * hh;
            float v = acc[r] + bias;
            out[(size_t)(b * TT + t0 + m) * TOUT + n] = v > 0.f ? v : 0.f;
        }
    }
}

// ---------------------------------------------------------------------------
extern "C" void kernel_launch(void* const* d_in, const int* in_sizes, int n_in,
                              void* d_out, int out_size, void* d_ws, size_t ws_size,
                              hipStream_t stream) {
    const float* x0    = (const float*)d_in[0];
    const float* x1    = (const float*)d_in[1];
    const float* projw = (const float*)d_in[2];
    const float* fcw   = (const float*)d_in[3];
    const float* fcb   = (const float*)d_in[4];
    float* out = (float*)d_out;

    // workspace layout (bf16 buffers)
    char* ws = (char*)d_ws;
    unsigned short* feats   = (unsigned short*)(ws);                                     // 4 MiB
    unsigned short* projw_t = (unsigned short*)(ws + (size_t)(4u << 20));                // 256 KiB
    unsigned short* fcw_t   = (unsigned short*)(ws + (size_t)(4u << 20) + (256u << 10)); // 32 KiB
    unsigned short* xn      = (unsigned short*)(ws + (size_t)(4u << 20) + (288u << 10)); // 512 KiB

    int convN = TIN * TSIM + TOUT * 128;
    k0_convert  <<<dim3((convN + 255) / 256), 256, 0, stream>>>(projw, fcw, projw_t, fcw_t);
    k1_pool     <<<dim3(TB * TIN, TT / POOL_TCHUNK), 256, 0, stream>>>(x0, x1, feats);
    k2_proj_norm<<<dim3(TB * TT / 16),        256, 0, stream>>>(feats, projw_t, xn);
    k3_sim_fc   <<<dim3(TB * TT / 16),        256, 0, stream>>>(xn, fcw_t, fcb, out);
}
